// IntraStripGCM_21577915695600
// MI455X (gfx1250) — compile-verified, matrix operand order checked by reading, not run on hardware
//
#include <hip/hip_runtime.h>

// Problem constants (from reference setup_inputs)
#define B_    4
#define C_    128
#define H_    64
#define W_    64
#define F_    8
#define NRES  4
#define LRELU 0.2f
#define LN_EPSF 1e-5f

typedef __attribute__((ext_vector_type(16))) _Float16 v16h;
typedef __attribute__((ext_vector_type(2)))  _Float16 v2h;
typedef __attribute__((ext_vector_type(8)))  float    v8f;

__device__ __forceinline__ float leaky(float v) { return v >= 0.f ? v : LRELU * v; }

__device__ __forceinline__ v2h splat2(float w) {
  _Float16 h = (_Float16)w;
  v2h r; r[0] = h; r[1] = h;
  return r;
}

// D = A(16x32 f16) x B(32x16 f16) + C(16x16 f32)  -> v_wmma_f32_16x16x32_f16
__device__ __forceinline__ v8f wmma_f16(v16h a, v16h b, v8f c) {
  return __builtin_amdgcn_wmma_f32_16x16x32_f16(false, a, false, b, (short)0, c,
                                                false, false);
}

// Load a 16x32 f16 fragment whose "row" dimension is striped across lanes and
// whose K dimension is row-contiguous in the f16 source (row-major, stride ld).
// Serves as the A loader (src indexed [M][K]) and the B loader for sources
// stored [N][K]. Lane layout (ISA 7.12.2, 16-bit A 16x32): e<8 -> K=half*8+e,
// e>=8 -> K=16+half*8+(e-8); both 8-element runs are contiguous 16B LDS loads.
__device__ __forceinline__ v16h load_frag16(const _Float16* __restrict__ src,
                                            int ld, int r0, int k0) {
  const int lane = threadIdx.x & 31;
  const int half = lane >> 4;
  const _Float16* p = src + (r0 + (lane & 15)) * ld + k0 + half * 8;
  v16h r;
#pragma unroll
  for (int e = 0; e < 8; ++e) r[e] = p[e];
#pragma unroll
  for (int e = 0; e < 8; ++e) r[8 + e] = p[16 + e];
  return r;
}

// ---------------------------------------------------------------------------
// Kernel A: LayerNorm(C) + 1x1 conv (WMMA GEMM) + leaky + split/scatter into
// horizontal/vertical strip layouts (f16 workspace).
//   fh: [B*H][64][W]   fv: [B*W][64][H]
// ---------------------------------------------------------------------------
__global__ void __launch_bounds__(256)
ln_conv_split_kernel(const float* __restrict__ x,
                     const float* __restrict__ norm_w,
                     const float* __restrict__ norm_b,
                     const float* __restrict__ conv_w,
                     const float* __restrict__ conv_b,
                     _Float16* __restrict__ fh, _Float16* __restrict__ fv) {
  __shared__ _Float16 wsm[C_ * C_];   // conv_w in f16 (32 KB)
  __shared__ float    xs[16 * C_];    // raw x tile [tok][c]
  __shared__ _Float16 xn[16 * C_];    // normalized f16 [tok][c]
  __shared__ float    red1[256], red2[256];
  __shared__ float    mu_s[16], rs_s[16];

  const int tid  = threadIdx.x;
  const int tok0 = blockIdx.x * 16;

  __builtin_prefetch(conv_w + tid * 16, 0, 1);   // global_prefetch_b8

  for (int i = tid; i < C_ * C_; i += 256) wsm[i] = (_Float16)conv_w[i];

  // Stage x tile, coalesced along token (w) dimension.
  for (int i = tid; i < 16 * C_; i += 256) {
    int c = i >> 4, t = i & 15;
    int token = tok0 + t;
    int b = token >> 12, rem = token & 4095;
    int hh = rem >> 6, ww = rem & 63;
    xs[t * C_ + c] = x[(((b * C_) + c) * H_ + hh) * W_ + ww];
  }
  __syncthreads();

  // LayerNorm: 16 threads per token, 8 channels each, tree-reduce in LDS.
  {
    int t = tid & 15, j = tid >> 4;
    float s = 0.f, s2 = 0.f;
#pragma unroll
    for (int cc = 0; cc < 8; ++cc) {
      float v = xs[t * C_ + j * 8 + cc];
      s += v; s2 += v * v;
    }
    red1[tid] = s; red2[tid] = s2;
  }
  __syncthreads();
  if (tid < 16) {
    float s = 0.f, s2 = 0.f;
#pragma unroll
    for (int j = 0; j < 16; ++j) { s += red1[j * 16 + tid]; s2 += red2[j * 16 + tid]; }
    float mu = s * (1.f / C_);
    float var = s2 * (1.f / C_) - mu * mu;
    mu_s[tid] = mu;
    rs_s[tid] = rsqrtf(var + LN_EPSF);
  }
  __syncthreads();
  {
    int t = tid & 15, j = tid >> 4;
    float mu = mu_s[t], rs = rs_s[t];
#pragma unroll
    for (int cc = 0; cc < 8; ++cc) {
      int c = j * 8 + cc;
      xn[t * C_ + c] = (_Float16)((xs[t * C_ + c] - mu) * rs * norm_w[c] + norm_b[c]);
    }
  }
  __syncthreads();

  // GEMM: each of 8 waves produces a 16(out-ch) x 16(token) tile, K=128.
  const int wave = tid >> 5;
  const int o0 = wave * 16;
  v8f acc = {};
#pragma unroll
  for (int k = 0; k < C_; k += 32)
    acc = wmma_f16(load_frag16(wsm, C_, o0, k), load_frag16(xn, C_, 0, k), acc);

  const int lane = tid & 31, half = lane >> 4;
  const int t = lane & 15;
  const int token = tok0 + t;
  const int b = token >> 12, rem = token & 4095;
  const int hh = rem >> 6, ww = rem & 63;
#pragma unroll
  for (int v = 0; v < 8; ++v) {
    int o = o0 + v + half * 8;                 // D layout: m = v + 8*half
    float val = leaky(acc[v] + conv_b[o]);
    if (o < 64) fh[(((b * H_) + hh) * 64 + o) * W_ + ww] = (_Float16)val;
    else        fv[(((b * W_) + ww) * 64 + (o - 64)) * H_ + hh] = (_Float16)val;
  }
}

// ---------------------------------------------------------------------------
// GCM kernel, BOTH directions in one launch (grid = 512: first 256 blocks are
// horizontal strips, last 256 vertical) so the two independent passes fill the
// machine together. One strip-graph per block (4 waves x 32). All adj
// applications are 64x64x64 f16 WMMA matmuls batched over 64 channels; the 8x8
// feature mixes run as packed-f16 VALU (v_pk_*). Tensors live in dynamic LDS
// as [f][c][l] so every WMMA fragment read is row-contiguous.
// ---------------------------------------------------------------------------
__global__ void __launch_bounds__(128)
gcm_dual_kernel(const _Float16* __restrict__ fh,   // [B*H][64][64]
                const _Float16* __restrict__ fv,   // [B*W][64][64]
                const float* __restrict__ adj_hg, const float* __restrict__ adj_vg,
                const float* __restrict__ head_hg, const float* __restrict__ w1_hg,
                const float* __restrict__ w2_hg,   const float* __restrict__ tail_hg,
                const float* __restrict__ head_vg, const float* __restrict__ w1_vg,
                const float* __restrict__ w2_vg,   const float* __restrict__ tail_vg,
                _Float16* __restrict__ attn) {     // [B][C][H][W]
  extern __shared__ char smem_raw[];
  _Float16* adj_s = (_Float16*)smem_raw;        //  4096
  _Float16* S_s   = adj_s + 4096;               //  4096  [c][l]
  _Float16* X_s   = S_s + 4096;                 // 32768  [f][c][l]
  _Float16* b1    = X_s + F_ * 4096;            // 32768
  _Float16* b2    = b1 + F_ * 4096;             // 32768
  float* wts      = (float*)(b2 + F_ * 4096);
  float* w1_s   = wts;
  float* w2_s   = wts + NRES * F_ * F_;
  float* head_s = w2_s + NRES * F_ * F_;
  float* tail_s = head_s + F_;

  const bool dirv = blockIdx.x >= (B_ * H_);
  const int  n    = dirv ? (blockIdx.x - B_ * H_) : blockIdx.x;

  const _Float16* feat   = dirv ? fv : fh;
  const float* adjg      = dirv ? adj_vg : adj_hg;
  const float* head_g    = dirv ? head_vg : head_hg;
  const float* w1_g      = dirv ? w1_vg : w1_hg;
  const float* w2_g      = dirv ? w2_vg : w2_hg;
  const float* tail_g    = dirv ? tail_vg : tail_hg;

  const int tid = threadIdx.x;
  const _Float16* fbase = feat + n * 4096;

  for (int i = tid; i < 4096; i += 128) {
    adj_s[i] = (_Float16)adjg[i];
    S_s[i]   = fbase[i];
  }
  for (int i = tid; i < NRES * F_ * F_; i += 128) {
    w1_s[i] = w1_g[i];
    w2_s[i] = w2_g[i];
  }
  if (tid < F_) { head_s[tid] = head_g[tid]; tail_s[tid] = tail_g[tid]; }
  __syncthreads();

  const int wave = tid >> 5;
  const int m0 = wave * 16;                 // this wave owns l-rows [m0, m0+16)
  const int lane = tid & 31, half = lane >> 4;

  // Head: T = adj @ S^T ; X[f][c][l] = head[f] * T[l][c]  (packed pair stores)
#pragma unroll
  for (int ct = 0; ct < 4; ++ct) {
    v8f acc = {};
    acc = wmma_f16(load_frag16(adj_s, 64, m0, 0),  load_frag16(S_s, 64, ct * 16, 0),  acc);
    acc = wmma_f16(load_frag16(adj_s, 64, m0, 32), load_frag16(S_s, 64, ct * 16, 32), acc);
    int c = ct * 16 + (lane & 15);
#pragma unroll
    for (int f = 0; f < F_; ++f) {
      float hw = head_s[f];
      v2h* Xp = (v2h*)(X_s + (f * 64 + c) * 64 + m0 + half * 8);
#pragma unroll
      for (int v = 0; v < 4; ++v) {
        v2h p; p[0] = (_Float16)(hw * acc[2 * v]); p[1] = (_Float16)(hw * acc[2 * v + 1]);
        Xp[v] = p;
      }
    }
  }
  __syncthreads();

  for (int it = 0; it < NRES; ++it) {
    const float* W1 = w1_s + it * F_ * F_;
    const float* W2 = w2_s + it * F_ * F_;

    // b1[f'][c][l] = sum_f X[f][c][l] * W1[f][f']   (packed f16 pairs)
    {
      const v2h* Xp = (const v2h*)X_s;
      v2h* Bp = (v2h*)b1;
      for (int i = tid; i < 2048; i += 128) {
        v2h xv[F_];
#pragma unroll
        for (int f = 0; f < F_; ++f) xv[f] = Xp[f * 2048 + i];
#pragma unroll
        for (int f2 = 0; f2 < F_; ++f2) {
          v2h s = splat2(0.f);
#pragma unroll
          for (int f = 0; f < F_; ++f) s += xv[f] * splat2(W1[f * F_ + f2]);
          Bp[f2 * 2048 + i] = s;
        }
      }
    }
    __syncthreads();

    // b2[f'] = leaky(adj @ b1[f'])  (packed pair stores)
    for (int f2 = 0; f2 < F_; ++f2) {
#pragma unroll
      for (int ct = 0; ct < 4; ++ct) {
        v8f acc = {};
        acc = wmma_f16(load_frag16(adj_s, 64, m0, 0),  load_frag16(b1 + f2 * 4096, 64, ct * 16, 0),  acc);
        acc = wmma_f16(load_frag16(adj_s, 64, m0, 32), load_frag16(b1 + f2 * 4096, 64, ct * 16, 32), acc);
        int c = ct * 16 + (lane & 15);
        v2h* Dp = (v2h*)(b2 + (f2 * 64 + c) * 64 + m0 + half * 8);
#pragma unroll
        for (int v = 0; v < 4; ++v) {
          v2h p; p[0] = (_Float16)leaky(acc[2 * v]); p[1] = (_Float16)leaky(acc[2 * v + 1]);
          Dp[v] = p;
        }
      }
    }
    __syncthreads();

    // b1[f'][c][l] = sum_f b2[f][c][l] * W2[f][f']   (packed f16 pairs)
    {
      const v2h* Rp = (const v2h*)b2;
      v2h* Bp = (v2h*)b1;
      for (int i = tid; i < 2048; i += 128) {
        v2h xv[F_];
#pragma unroll
        for (int f = 0; f < F_; ++f) xv[f] = Rp[f * 2048 + i];
#pragma unroll
        for (int f2 = 0; f2 < F_; ++f2) {
          v2h s = splat2(0.f);
#pragma unroll
          for (int f = 0; f < F_; ++f) s += xv[f] * splat2(W2[f * F_ + f2]);
          Bp[f2 * 2048 + i] = s;
        }
      }
    }
    __syncthreads();

    // X += adj @ b1  (packed RMW; each (f,c,l) pair owned by one lane)
    for (int f2 = 0; f2 < F_; ++f2) {
#pragma unroll
      for (int ct = 0; ct < 4; ++ct) {
        v8f acc = {};
        acc = wmma_f16(load_frag16(adj_s, 64, m0, 0),  load_frag16(b1 + f2 * 4096, 64, ct * 16, 0),  acc);
        acc = wmma_f16(load_frag16(adj_s, 64, m0, 32), load_frag16(b1 + f2 * 4096, 64, ct * 16, 32), acc);
        int c = ct * 16 + (lane & 15);
        v2h* Xp = (v2h*)(X_s + (f2 * 64 + c) * 64 + m0 + half * 8);
#pragma unroll
        for (int v = 0; v < 4; ++v) {
          v2h p; p[0] = (_Float16)acc[2 * v]; p[1] = (_Float16)acc[2 * v + 1];
          Xp[v] = Xp[v] + p;
        }
      }
    }
    __syncthreads();
  }

  // Tail: t[c][l] = sum_f X[f][c][l] * tail[f]  -> b1 (packed)
  {
    const v2h* Xp = (const v2h*)X_s;
    v2h* Bp = (v2h*)b1;
    for (int i = tid; i < 2048; i += 128) {
      v2h s = splat2(0.f);
#pragma unroll
      for (int f = 0; f < F_; ++f) s += Xp[f * 2048 + i] * splat2(tail_s[f]);
      Bp[i] = s;
    }
  }
  __syncthreads();

  // out = leaky(adj @ t) ; scatter into attn[b][c][h][w]
  const int bb  = n >> 6;   // H_ == W_ == 64
  const int pos = n & 63;
#pragma unroll
  for (int ct = 0; ct < 4; ++ct) {
    v8f acc = {};
    acc = wmma_f16(load_frag16(adj_s, 64, m0, 0),  load_frag16(b1, 64, ct * 16, 0),  acc);
    acc = wmma_f16(load_frag16(adj_s, 64, m0, 32), load_frag16(b1, 64, ct * 16, 32), acc);
    int c = ct * 16 + (lane & 15);
#pragma unroll
    for (int v = 0; v < 8; ++v) {
      int l = m0 + v + half * 8;
      float val = leaky(acc[v]);
      int idx;
      if (!dirv) idx = (((bb * C_) + c) * H_ + pos) * W_ + l;        // h-dir: strip=(b,h), l=w
      else       idx = (((bb * C_) + 64 + c) * H_ + l) * W_ + pos;   // v-dir: strip=(b,w), l=h
      attn[idx] = (_Float16)val;
    }
  }
}

// ---------------------------------------------------------------------------
// Kernel D: fuse GEMM (WMMA) + bias + residual add -> f32 output.
// ---------------------------------------------------------------------------
__global__ void __launch_bounds__(256)
fuse_res_kernel(const _Float16* __restrict__ attn,
                const float* __restrict__ fuse_w,
                const float* __restrict__ fuse_b,
                const float* __restrict__ x, float* __restrict__ out) {
  __shared__ _Float16 wsm[C_ * C_];
  __shared__ _Float16 as[16 * C_];   // [tok][c]
  const int tid = threadIdx.x;
  const int tok0 = blockIdx.x * 16;

  __builtin_prefetch(fuse_w + tid * 16, 0, 1);

  for (int i = tid; i < C_ * C_; i += 256) wsm[i] = (_Float16)fuse_w[i];
  for (int i = tid; i < 16 * C_; i += 256) {
    int c = i >> 4, t = i & 15;
    int token = tok0 + t;
    int b = token >> 12, rem = token & 4095;
    int hh = rem >> 6, ww = rem & 63;
    as[t * C_ + c] = attn[(((b * C_) + c) * H_ + hh) * W_ + ww];
  }
  __syncthreads();

  const int wave = tid >> 5, o0 = wave * 16;
  v8f acc = {};
#pragma unroll
  for (int k = 0; k < C_; k += 32)
    acc = wmma_f16(load_frag16(wsm, C_, o0, k), load_frag16(as, C_, 0, k), acc);

  const int lane = tid & 31, half = lane >> 4, t = lane & 15;
  const int token = tok0 + t;
  const int b = token >> 12, rem = token & 4095;
  const int hh = rem >> 6, ww = rem & 63;
#pragma unroll
  for (int v = 0; v < 8; ++v) {
    int o = o0 + v + half * 8;
    int gi = (((b * C_) + o) * H_ + hh) * W_ + ww;
    out[gi] = acc[v] + fuse_b[o] + x[gi];
  }
}

// ---------------------------------------------------------------------------
extern "C" void kernel_launch(void* const* d_in, const int* in_sizes, int n_in,
                              void* d_out, int out_size, void* d_ws, size_t ws_size,
                              hipStream_t stream) {
  (void)in_sizes; (void)n_in; (void)out_size; (void)ws_size;
  const float* x        = (const float*)d_in[0];
  const float* adj_h    = (const float*)d_in[1];
  const float* adj_v    = (const float*)d_in[2];
  const float* norm_w   = (const float*)d_in[3];
  const float* norm_b   = (const float*)d_in[4];
  const float* conv_w   = (const float*)d_in[5];
  const float* conv_b   = (const float*)d_in[6];
  const float* fuse_w   = (const float*)d_in[7];
  const float* fuse_b   = (const float*)d_in[8];
  const float* head_h   = (const float*)d_in[9];
  const float* tail_h   = (const float*)d_in[10];
  const float* res_h_w1 = (const float*)d_in[11];
  const float* res_h_w2 = (const float*)d_in[12];
  const float* head_v   = (const float*)d_in[13];
  const float* tail_v   = (const float*)d_in[14];
  const float* res_v_w1 = (const float*)d_in[15];
  const float* res_v_w2 = (const float*)d_in[16];
  float* outp = (float*)d_out;

  char* ws = (char*)d_ws;
  _Float16* fh   = (_Float16*)(ws);                    // B*H*64*W f16 = 2 MB
  _Float16* fv   = (_Float16*)(ws + (2u << 20));       // B*W*64*H f16 = 2 MB
  _Float16* attn = (_Float16*)(ws + (4u << 20));       // B*C*H*W  f16 = 4 MB

  const int tokens = B_ * H_ * W_;   // 16384

  ln_conv_split_kernel<<<tokens / 16, 256, 0, stream>>>(x, norm_w, norm_b,
                                                        conv_w, conv_b, fh, fv);

  const size_t gcm_lds =
      (size_t)(2 * 4096 + 3 * F_ * 4096) * sizeof(_Float16) +
      (size_t)(2 * NRES * F_ * F_ + 2 * F_) * sizeof(float);   // ~215 KB < 320 KB WGP LDS

  gcm_dual_kernel<<<B_ * H_ + B_ * W_, 128, gcm_lds, stream>>>(
      fh, fv, adj_h, adj_v,
      head_h, res_h_w1, res_h_w2, tail_h,
      head_v, res_v_w1, res_v_w2, tail_v, attn);

  fuse_res_kernel<<<tokens / 16, 256, 0, stream>>>(attn, fuse_w, fuse_b, x, outp);
}
